// Gcn_22127671509489
// MI455X (gfx1250) — matrix operand order repaired and checked by reading, not compiled, not run on hardware
//
#include <hip/hip_runtime.h>

// ---- ext-vector types (trivial: safe in unions / reinterpret loads) ----
typedef __attribute__((ext_vector_type(16))) __bf16       v16bf;
typedef __attribute__((ext_vector_type(8)))  float        v8f;
typedef __attribute__((ext_vector_type(4)))  float        f32x4;
typedef __attribute__((ext_vector_type(4)))  unsigned int u32x4;

// =====================================================================
// Degree + norm kernels
// =====================================================================
__global__ void k_degrees(const int* __restrict__ src, const int* __restrict__ dst,
                          float* __restrict__ outdeg, float* __restrict__ indeg, int E) {
  int i = blockIdx.x * blockDim.x + threadIdx.x;
  if (i < E) {
    atomicAdd(&outdeg[src[i]], 1.0f);
    atomicAdd(&indeg[dst[i]], 1.0f);
  }
}

__global__ void k_norms(float* __restrict__ a, float* __restrict__ b, int N) {
  int i = blockIdx.x * blockDim.x + threadIdx.x;
  if (i < N) {
    float x = a[i];
    a[i] = (x > 0.0f) ? rsqrtf(x) : 0.0f;
    float y = b[i];
    b[i] = (y > 0.0f) ? rsqrtf(y) : 0.0f;
  }
}

// =====================================================================
// Edge aggregation: agg[dst] += h[src] * ns[src]   (float4-vectorized)
// d4_shift = log2(d/4). Mostly L2-resident atomics (node matrix < 192MB L2).
// =====================================================================
__global__ void k_aggregate(const float* __restrict__ h, const int* __restrict__ src,
                            const int* __restrict__ dst, const float* __restrict__ ns,
                            float* __restrict__ agg, int d4_shift, long long total) {
  long long i = (long long)blockIdx.x * blockDim.x + threadIdx.x;
  if (i >= total) return;
  int d4m = (1 << d4_shift) - 1;
  int e = (int)(i >> d4_shift);
  int f = (int)(i & d4m);
  int s = src[e], t = dst[e];
  float sc = ns[s];
  f32x4 v = reinterpret_cast<const f32x4*>(h)[((size_t)s << d4_shift) + f];
  float* o = agg + ((((size_t)t << d4_shift) + f) << 2);
  atomicAdd(o + 0, v[0] * sc);
  atomicAdd(o + 1, v[1] * sc);
  atomicAdd(o + 2, v[2] * sc);
  atomicAdd(o + 3, v[3] * sc);
}

// =====================================================================
// Weight transpose + fp32->bf16:  Wt[n*K + k] = bf16(W[k*ncol + n])
// =====================================================================
__global__ void k_wconv(const float* __restrict__ W, __bf16* __restrict__ Wt,
                        int K, int ncol) {
  int i = blockIdx.x * blockDim.x + threadIdx.x;
  if (i >= K * ncol) return;
  int n = i % ncol, k = i / ncol;
  Wt[(size_t)n * K + k] = (__bf16)W[(size_t)k * ncol + n];
}

// =====================================================================
// Activation convert: Abf[r*K+k] = bf16(A[r*K+k] * nd[r])   (8-wide)
// k8_shift = log2(K/8); total = N*K/8
// =====================================================================
__global__ void k_aconv(const float* __restrict__ A, const float* __restrict__ ndp,
                        __bf16* __restrict__ Abf, int k8_shift, long long total) {
  long long i = (long long)blockIdx.x * blockDim.x + threadIdx.x;
  if (i >= total) return;
  int r  = (int)(i >> k8_shift);
  int kc = (int)(i & ((1 << k8_shift) - 1)) << 3;     // element offset (8-wide)
  float s = ndp ? ndp[r] : 1.0f;
  size_t base = ((size_t)r << (k8_shift + 3)) + kc;
  const f32x4* p = reinterpret_cast<const f32x4*>(A + base);
  f32x4 x = p[0], y = p[1];
  union { __bf16 h[8]; u32x4 q; } o;
#pragma unroll
  for (int j = 0; j < 4; ++j) {
    o.h[j]     = (__bf16)(x[j] * s);
    o.h[4 + j] = (__bf16)(y[j] * s);
  }
  *reinterpret_cast<u32x4*>(Abf + base) = o.q;
}

// =====================================================================
// WMMA GEMM:  Out = act( Abf @ W + bias )
//   Abf : [Nrows x K] bf16 row-major (nd already folded in)
//   Wt  : [ncol x K]  bf16 (pre-transposed column panel; contiguous in K)
//   Wave: one 16-col tile x 5 M-subtiles (80 rows); B-frag reused 5x.
//   Block = 8 waves = 128 cols. K % 32 == 0 required.
// =====================================================================
#define MT 5
__global__ __launch_bounds__(256)
void k_gemm_bf16(const __bf16* __restrict__ Abf, const __bf16* __restrict__ Wt,
                 const float* __restrict__ bias, float* __restrict__ Out,
                 int Nrows, int K, int ncol, int relu) {
  const int wave = threadIdx.x >> 5;
  const int lane = threadIdx.x & 31;
  const int l16  = lane & 15;
  const int hi   = lane >> 4;                 // 0: lanes 0-15, 1: lanes 16-31
  const int ct   = blockIdx.y * 8 + wave;     // 16-col tile index
  const int colbase = ct * 16;
  if (colbase >= ncol) return;                // uniform per wave -> EXEC stays full
  const int rowbase = blockIdx.x * (MT * 16);

  const int n = colbase + l16;                // this lane's output column
  const float bv = bias ? bias[n] : 0.0f;
  const __bf16* wrow = Wt + (size_t)n * K;

  const __bf16* arow[MT];
#pragma unroll
  for (int t = 0; t < MT; ++t) {
    int ra = rowbase + t * 16 + l16;
    if (ra >= Nrows) ra = Nrows - 1;          // clamp (loads only; stores guarded)
    arow[t] = Abf + (size_t)ra * K;
  }

  v8f acc[MT];
#pragma unroll
  for (int t = 0; t < MT; ++t)
    acc[t] = (v8f){0.f, 0.f, 0.f, 0.f, 0.f, 0.f, 0.f, 0.f};

  const int offA = hi * 16;   // A k order chosen here (matches B): contiguous 16-wide K per half-wave
  const int offB = hi * 16;   // B k order chosen here (matches A): contiguous 16-wide K per half-wave

#pragma unroll 2
  for (int k0 = 0; k0 < K; k0 += 32) {
    // B fragment: 16 contiguous bf16 (32B aligned)
    v16bf bfrag = *reinterpret_cast<const v16bf*>(wrow + k0 + offB);

    // Issue all A-fragment loads up front (independent), then 5 WMMAs.
    union AF { u32x4 q[2]; v16bf v; } a[MT];
#pragma unroll
    for (int t = 0; t < MT; ++t) {
      a[t].q[0] = *reinterpret_cast<const u32x4*>(arow[t] + k0 + offA);
      a[t].q[1] = *reinterpret_cast<const u32x4*>(arow[t] + k0 + offA + 8);
    }
#pragma unroll
    for (int t = 0; t < MT; ++t) {
      acc[t] = __builtin_amdgcn_wmma_f32_16x16x32_bf16(
          false, a[t].v, false, bfrag, (short)0, acc[t], false, false);
    }
  }

  // Epilogue: C/D layout -> VGPR r holds (M = r + hi*8, N = l16)
#pragma unroll
  for (int t = 0; t < MT; ++t) {
#pragma unroll
    for (int r = 0; r < 8; ++r) {
      int row = rowbase + t * 16 + hi * 8 + r;
      if (row < Nrows) {
        float v = acc[t][r] + bv;
        if (relu) v = (v > 0.0f) ? v : 0.0f;
        Out[(size_t)row * ncol + n] = v;
      }
    }
  }
}

// =====================================================================
// Final epilogue (layer 3): out = agg64 * nd[row] + b3[col]
// =====================================================================
__global__ void k_final(const float* __restrict__ agg, const float* __restrict__ nd,
                        const float* __restrict__ b, float* __restrict__ out,
                        int N, int C) {
  int i = blockIdx.x * blockDim.x + threadIdx.x;
  if (i >= N * C) return;
  int n = i / C, c = i % C;
  out[i] = agg[i] * nd[n] + b[c];
}

// =====================================================================
// Launch
// =====================================================================
extern "C" void kernel_launch(void* const* d_in, const int* in_sizes, int n_in,
                              void* d_out, int out_size, void* d_ws, size_t ws_size,
                              hipStream_t stream) {
  const float* feat = (const float*)d_in[0];
  const int*   src  = (const int*)d_in[1];
  const int*   dst  = (const int*)d_in[2];
  const float* W1   = (const float*)d_in[3];
  const float* b1   = (const float*)d_in[4];
  const float* W2   = (const float*)d_in[5];
  const float* b2   = (const float*)d_in[6];
  const float* W3   = (const float*)d_in[7];
  const float* b3   = (const float*)d_in[8];
  float* out = (float*)d_out;

  const int H = in_sizes[4];            // 512
  const int F = in_sizes[3] / H;        // 512
  const int C = in_sizes[8];            // 64
  const int N = in_sizes[0] / F;        // 50000
  const int E = in_sizes[1];            // 1600000

  // ---- workspace carve-up (256B aligned) ----
  char* w = (char*)d_ws;
  auto aln = [](size_t x) { return (x + 255) & ~(size_t)255; };
  float*  ns   = (float*)w;  w += aln((size_t)N * 4);
  float*  nd   = (float*)w;  w += aln((size_t)N * 4);
  __bf16* Wt   = (__bf16*)w; w += aln((size_t)F * H * 2);     // largest weight
  __bf16* Abf  = (__bf16*)w; w += aln((size_t)N * H * 2);     // bf16 activations
  float*  bufA = (float*)w;  w += aln((size_t)N * H * 4);
  float*  bufB = (float*)w;  w += aln((size_t)N * H * 4);
  (void)ws_size; (void)n_in; (void)out_size;

  const int TPB = 256;
  const int shF  = __builtin_ctz(F / 4);   // aggregate shift (F)
  const int shH  = __builtin_ctz(H / 4);   // aggregate shift (H)
  const int shC  = __builtin_ctz(C / 4);   // aggregate shift (C)
  const int sh8F = __builtin_ctz(F / 8);   // aconv shift (K=F)
  const int sh8H = __builtin_ctz(H / 8);   // aconv shift (K=H)

  // ---- degrees / norms ----
  hipMemsetAsync(ns, 0, (size_t)N * 4, stream);
  hipMemsetAsync(nd, 0, (size_t)N * 4, stream);
  k_degrees<<<(E + TPB - 1) / TPB, TPB, 0, stream>>>(src, dst, ns, nd, E);
  k_norms<<<(N + TPB - 1) / TPB, TPB, 0, stream>>>(ns, nd, N);

  dim3 gridHH((N + MT * 16 - 1) / (MT * 16), (H + 127) / 128);
  dim3 gridHC((N + MT * 16 - 1) / (MT * 16), (C + 127) / 128);
  long long totCvt = (long long)N * (H / 8);
  unsigned  cvtBlocks = (unsigned)((totCvt + TPB - 1) / TPB);

  // ---- layer 1: agg(features) -> bufA ; h1 = relu(agg*nd @ W1 + b1) -> bufB ----
  hipMemsetAsync(bufA, 0, (size_t)N * H * 4, stream);
  {
    long long tot = (long long)E * (F / 4);
    k_aggregate<<<(unsigned)((tot + TPB - 1) / TPB), TPB, 0, stream>>>(
        feat, src, dst, ns, bufA, shF, tot);
  }
  k_aconv<<<cvtBlocks, TPB, 0, stream>>>(bufA, nd, Abf, sh8F, (long long)N * (F / 8));
  k_wconv<<<(F * H + TPB - 1) / TPB, TPB, 0, stream>>>(W1, Wt, F, H);
  k_gemm_bf16<<<gridHH, TPB, 0, stream>>>(Abf, Wt, b1, bufB, N, F, H, 1);

  // ---- layer 2: agg(h1) -> bufA ; h2 = relu(agg*nd @ W2 + b2) -> bufB ----
  hipMemsetAsync(bufA, 0, (size_t)N * H * 4, stream);
  {
    long long tot = (long long)E * (H / 4);
    k_aggregate<<<(unsigned)((tot + TPB - 1) / TPB), TPB, 0, stream>>>(
        bufB, src, dst, ns, bufA, shH, tot);
  }
  k_aconv<<<cvtBlocks, TPB, 0, stream>>>(bufA, nd, Abf, sh8H, (long long)N * (H / 8));
  k_wconv<<<(H * H + TPB - 1) / TPB, TPB, 0, stream>>>(W2, Wt, H, H);
  k_gemm_bf16<<<gridHH, TPB, 0, stream>>>(Abf, Wt, b2, bufB, N, H, H, 1);

  // ---- layer 3 (GEMM-first reorder): t3 = h2 @ W3 -> bufA[:,0:C] ----
  k_aconv<<<cvtBlocks, TPB, 0, stream>>>(bufB, nullptr, Abf, sh8H, (long long)N * (H / 8));
  k_wconv<<<(H * C + TPB - 1) / TPB, TPB, 0, stream>>>(W3, Wt, H, C);
  k_gemm_bf16<<<gridHC, TPB, 0, stream>>>(Abf, Wt, nullptr, bufA, N, H, C, 0);

  // agg64(t3) -> bufB[:,0:C] ; out = agg64 * nd + b3
  hipMemsetAsync(bufB, 0, (size_t)N * C * 4, stream);
  {
    long long tot = (long long)E * (C / 4);
    k_aggregate<<<(unsigned)((tot + TPB - 1) / TPB), TPB, 0, stream>>>(
        bufA, src, dst, ns, bufB, shC, tot);
  }
  k_final<<<(N * C + TPB - 1) / TPB, TPB, 0, stream>>>(bufB, nd, b3, out, N, C);
}
